// Model_76304388980798
// MI455X (gfx1250) — compile-verified, matrix operand order checked by reading, not run on hardware
//
#include <hip/hip_runtime.h>
#include <math.h>

// ---------------------------------------------------------------------------
// Transformer encoder-decoder forward (MI455X / gfx1250, wave32).
// GEMMs: v_wmma_f32_16x16x32_f16, A operand kept as f16 in memory and staged
// to LDS with GLOBAL_LOAD_ASYNC_TO_LDS_B128 (ASYNCcnt), double-buffered so the
// async DMA and B-tile staging overlap the WMMA work of the current tile.
// Attention is fused flash-style (1 wave / query). Classifier GEMM is chunked
// (N tiles of 2048) with streaming logsumexp so the 16x256x32000 logits are
// never fully materialized.
// ---------------------------------------------------------------------------

#define D_   512
#define H_   8
#define DH_  64
#define F_   2048
#define L_   6
#define B_   16
#define SE_  256
#define SC_  256
#define VC_  32000
#define EPS_ 1e-5f

typedef __attribute__((ext_vector_type(16))) _Float16 v16h;
typedef __attribute__((ext_vector_type(8)))  float    v8f;

// Async global -> LDS 16-byte copy (GVS mode: saddr base + 32-bit voffset).
// LDS aperture lives in addr[63:32] (ISA 10.2), so the low 32 bits of a
// __shared__ pointer are the LDS byte address the VDST operand expects.
__device__ __forceinline__ void async_ld_b128(unsigned lds_addr, unsigned gvoff,
                                              const void* base)
{
  asm volatile("global_load_async_to_lds_b128 %0, %1, %2"
               :
               : "v"(lds_addr), "v"(gvoff),
                 "s"((unsigned long long)(size_t)base)
               : "memory");
}
__device__ __forceinline__ void wait_asynccnt0()
{
  asm volatile("s_wait_asynccnt 0x0" ::: "memory");
}

// ---------------------------------------------------------------------------
// WMMA GEMM:  C[M,N] = A[M,K](f16) @ W[K,N](f32->f16) (+bias) (optional ReLU)
// Block tile 128x128, K-step 32, 256 threads = 8 waves (2x4 wave grid),
// each wave computes 4x2 tiles of 16x16. Double-buffered LDS; A tiles arrive
// via async-to-LDS DMA, W tiles are converted f32->f16 and transposed into
// LDS by stores. M, N multiples of 128; K multiple of 32 (true here).
// ---------------------------------------------------------------------------
__global__ __launch_bounds__(256) void gemm_wmma(
    const _Float16* __restrict__ A, const float* __restrict__ W,
    const float* __restrict__ bias, float* __restrict__ C,
    int M, int N, int K, int ldw, int ldc, int relu)
{
  __shared__ _Float16 As[2][128][40];  // [buf][row][k], +8 half pad (80B rows)
  __shared__ _Float16 Bs[2][128][40];  // [buf][col][k] (transposed staging)

  const int tid  = threadIdx.x;
  const int lane = tid & 31;
  const int wave = tid >> 5;
  const int wr   = wave >> 2;          // 0..1 -> 64-row slab
  const int wc   = wave & 3;           // 0..3 -> 32-col slab
  const int m0   = blockIdx.y * 128;
  const int n0   = blockIdx.x * 128;
  const int lrow  = lane & 15;
  const int kbase = (lane < 16) ? 0 : 8;  // ISA 16-bit A/B K packing

  v8f acc[4][2] = {};

  // A staging: 512 16-byte chunks (128 rows x 4), 2 per thread, async DMA.
  const int ar0 = tid >> 1;                  // chunk0: row, seg
  const int ac0 = (tid & 1) * 2;             //   segs 0..1 or 2..3 -> use 2 chunks
  // B staging: 32 k-rows x 8 col-segments of 16.
  const int bk  = tid >> 3;
  const int bns = (tid & 7) * 16;

  auto stageA = [&](int p, int k0) {
    #pragma unroll
    for (int c = 0; c < 2; ++c) {
      const int r  = ar0;
      const int ch = ac0 + c;                // 16B chunk within the 64B row
      const unsigned lds = (unsigned)(size_t)&As[p][r][ch * 8];
      const unsigned gof =
          (unsigned)(((size_t)(m0 + r) * K + k0 + ch * 8) * sizeof(_Float16));
      async_ld_b128(lds, gof, A);
    }
  };
  auto stageB = [&](int p, int k0) {
    const float* wp = W + (size_t)(k0 + bk) * ldw + n0 + bns;
    #pragma unroll
    for (int j = 0; j < 16; ++j) Bs[p][bns + j][bk] = (_Float16)wp[j];
    if (k0 + 32 < K)  // pull next W tile toward L2/WGP$ while we compute
      __builtin_prefetch(W + (size_t)(k0 + 32 + bk) * ldw + n0 + bns, 0, 1);
  };

  const int nk = K >> 5;
  stageA(0, 0);
  stageB(0, 0);

  for (int kt = 0; kt < nk; ++kt) {
    const int p = kt & 1;
    wait_asynccnt0();          // my async DMAs for buffer p have landed
    __syncthreads();           // everyone's DMAs/stores for p visible;
                               // also closes WAR on buffer p^1
    if (kt + 1 < nk) {         // overlap next tile's DMA + stores with WMMA
      stageA(p ^ 1, (kt + 1) << 5);
      stageB(p ^ 1, (kt + 1) << 5);
    }

    v16h bf[2];
    #pragma unroll
    for (int nc = 0; nc < 2; ++nc) {
      const _Float16* pb = &Bs[p][wc * 32 + nc * 16 + lrow][kbase];
      #pragma unroll
      for (int i = 0; i < 8; ++i) { bf[nc][i] = pb[i]; bf[nc][8 + i] = pb[16 + i]; }
    }
    #pragma unroll
    for (int mr = 0; mr < 4; ++mr) {
      const _Float16* pa = &As[p][wr * 64 + mr * 16 + lrow][kbase];
      v16h af;
      #pragma unroll
      for (int i = 0; i < 8; ++i) { af[i] = pa[i]; af[8 + i] = pa[16 + i]; }
      acc[mr][0] = __builtin_amdgcn_wmma_f32_16x16x32_f16(
          false, af, false, bf[0], (short)0, acc[mr][0], false, false);
      acc[mr][1] = __builtin_amdgcn_wmma_f32_16x16x32_f16(
          false, af, false, bf[1], (short)0, acc[mr][1], false, false);
    }
  }

  // C/D layout: VGPR r -> M = r (lanes 0-15) / r+8 (lanes 16-31); N = lane%16
  const int radd = (lane < 16) ? 0 : 8;
  #pragma unroll
  for (int mr = 0; mr < 4; ++mr) {
    #pragma unroll
    for (int nc = 0; nc < 2; ++nc) {
      const int col = n0 + wc * 32 + nc * 16 + lrow;
      const float bv = bias ? bias[col] : 0.f;
      #pragma unroll
      for (int r = 0; r < 8; ++r) {
        const int row = m0 + wr * 64 + mr * 16 + r + radd;
        float v = acc[mr][nc][r] + bv;
        if (relu) v = fmaxf(v, 0.f);
        C[(size_t)row * ldc + col] = v;
      }
    }
  }
}

// ---------------------------------------------------------------------------
// f32 -> f16 convert (for the GEMM A operand mirror)
// ---------------------------------------------------------------------------
__global__ __launch_bounds__(256) void cvt_f32_f16(
    const float* __restrict__ in, _Float16* __restrict__ out, int n)
{
  const int stride = gridDim.x * 256;
  for (int i = blockIdx.x * 256 + threadIdx.x; i < n; i += stride)
    out[i] = (_Float16)in[i];
}

// ---------------------------------------------------------------------------
// Embedding + sinusoidal positional encoding (+ length mask on embeddings)
// ---------------------------------------------------------------------------
__global__ __launch_bounds__(256) void embed_pe(
    const int* __restrict__ toks, const int* __restrict__ lens,
    const float* __restrict__ emb, float* __restrict__ out, int S)
{
  const int row = blockIdx.x;
  const int b = row / S, s = row % S;
  const int tok = toks[row];
  const int len = lens[b];
  const float t = (float)(s + 1);
  for (int i = threadIdx.x; i < D_; i += 256) {
    const int j = i & ~1;
    const float wi  = __powf(10000.0f, -(float)j / (float)D_);
    const float ang = t * wi;
    const float pe  = (i & 1) ? __cosf(ang) : __sinf(ang);
    const float e   = (s < len) ? emb[(size_t)tok * D_ + i] : 0.0f;
    out[(size_t)row * D_ + i] = e + pe;
  }
}

// ---------------------------------------------------------------------------
// LayerNorm(tmp)*g + lb + res -> out   (out may alias res; elementwise)
// ---------------------------------------------------------------------------
__global__ __launch_bounds__(256) void ln_add(
    const float* __restrict__ t, const float* __restrict__ g,
    const float* __restrict__ lb, const float* __restrict__ res,
    float* __restrict__ out)
{
  __shared__ float red[256];
  const size_t base = (size_t)blockIdx.x * D_;
  const int tid = threadIdx.x;
  const float x0 = t[base + tid], x1 = t[base + tid + 256];
  red[tid] = x0 + x1; __syncthreads();
  for (int s = 128; s > 0; s >>= 1) { if (tid < s) red[tid] += red[tid + s]; __syncthreads(); }
  const float mu = red[0] * (1.0f / D_);
  __syncthreads();
  const float d0 = x0 - mu, d1 = x1 - mu;
  red[tid] = d0 * d0 + d1 * d1; __syncthreads();
  for (int s = 128; s > 0; s >>= 1) { if (tid < s) red[tid] += red[tid + s]; __syncthreads(); }
  const float rs = rsqrtf(red[0] * (1.0f / D_) + EPS_);
  const float r0 = res[base + tid], r1 = res[base + tid + 256];
  out[base + tid]       = d0 * rs * g[tid]       + lb[tid]       + r0;
  out[base + tid + 256] = d1 * rs * g[tid + 256] + lb[tid + 256] + r1;
}

// ---------------------------------------------------------------------------
// Fused attention: one wave32 per (b, h, q) row. dh=64 -> 2 floats/lane.
// Masked keys (>= len, or causal-future) underflow to exactly 0 in the
// reference (exp(-1e9)), so they become loop bounds here.
// ---------------------------------------------------------------------------
__global__ __launch_bounds__(256) void attn_fused(
    const float* __restrict__ Qm, const float* __restrict__ Km,
    const float* __restrict__ Vm, float* __restrict__ Om,
    const int* __restrict__ lens, int Sq, int Sk, int causal)
{
  const int lane = threadIdx.x & 31;
  const int wave = threadIdx.x >> 5;
  const int gq = blockIdx.x * 8 + wave;
  const int b  = gq / (H_ * Sq);
  const int r  = gq % (H_ * Sq);
  const int h  = r / Sq;
  const int qi = r % Sq;
  const int len = lens[b];

  const float* qp = Qm + ((size_t)(b * Sq + qi) * D_ + h * DH_ + lane * 2);
  const float q0 = qp[0], q1 = qp[1];

  int kend = (len < Sk) ? len : Sk;
  if (causal && qi + 1 < kend) kend = qi + 1;   // len >= 1 so kend >= 1

  float m = -INFINITY, l = 0.f, a0 = 0.f, a1 = 0.f;
  for (int kk = 0; kk < kend; ++kk) {
    const size_t kb = (size_t)(b * Sk + kk) * D_ + h * DH_ + lane * 2;
    float part = q0 * Km[kb] + q1 * Km[kb + 1];
    #pragma unroll
    for (int off = 16; off > 0; off >>= 1) part += __shfl_xor(part, off, 32);
    const float s  = part * 0.125f;            // 1/sqrt(64)
    const float mn = fmaxf(m, s);
    const float corr = __expf(m - mn);         // 0 on first iter (m=-inf)
    const float p    = __expf(s - mn);
    l  = l * corr + p;
    a0 = a0 * corr + p * Vm[kb];
    a1 = a1 * corr + p * Vm[kb + 1];
    m = mn;
  }
  const float inv = 1.0f / l;
  float* op = Om + ((size_t)(b * Sq + qi) * D_ + h * DH_ + lane * 2);
  op[0] = a0 * inv;
  op[1] = a1 * inv;
}

// ---------------------------------------------------------------------------
// Streaming logsumexp state: st[row] = {running max, running sumexp, z_label}
// ---------------------------------------------------------------------------
__global__ void init_state(float* __restrict__ st)
{
  const int i = blockIdx.x * blockDim.x + threadIdx.x;
  if (i < B_ * SC_) { st[i * 3] = -INFINITY; st[i * 3 + 1] = 0.f; st[i * 3 + 2] = 0.f; }
}

__global__ __launch_bounds__(256) void lse_update(
    const float* __restrict__ logits, float* __restrict__ st,
    const int* __restrict__ chi, int ncols, int col0, int ldc)
{
  __shared__ float red[256];
  const int tid = threadIdx.x;
  const int row = blockIdx.x;
  const int b = row / SC_, t = row % SC_;
  const int lbl = (t < SC_ - 1) ? chi[b * SC_ + t + 1] : -1;

  float tm = -INFINITY, ts = 0.f, tz = 0.f;
  int found = 0;
  for (int c = tid; c < ncols; c += 256) {
    const float v = logits[(size_t)row * ldc + c];
    if (v > tm) { ts *= __expf(tm - v); tm = v; }
    ts += __expf(v - tm);
    if (col0 + c == lbl) { tz = v; found = 1; }
  }
  red[tid] = tm; __syncthreads();
  for (int s = 128; s > 0; s >>= 1) { if (tid < s) red[tid] = fmaxf(red[tid], red[tid + s]); __syncthreads(); }
  const float gm = red[0]; __syncthreads();
  red[tid] = ts * __expf(tm - gm); __syncthreads();
  for (int s = 128; s > 0; s >>= 1) { if (tid < s) red[tid] += red[tid + s]; __syncthreads(); }
  if (tid == 0) {
    const float gs = red[0];
    const float rm = st[row * 3], rs = st[row * 3 + 1];
    const float nm = fmaxf(rm, gm);
    const float ns = rs * __expf(rm - nm) + gs * __expf(gm - nm);
    st[row * 3] = nm; st[row * 3 + 1] = ns;
  }
  if (found) st[row * 3 + 2] = tz;
}

__global__ __launch_bounds__(256) void nll_mean(const float* __restrict__ st,
                                                float* __restrict__ out)
{
  __shared__ float red[256];
  const int tid = threadIdx.x;
  float s = 0.f;
  for (int row = tid; row < B_ * SC_; row += 256) {
    const int t = row % SC_;
    if (t >= SC_ - 1) continue;            // logits[:, :-1]
    const float lse = st[row * 3] + __logf(st[row * 3 + 1]);
    s += lse - st[row * 3 + 2];
  }
  red[tid] = s; __syncthreads();
  for (int k = 128; k > 0; k >>= 1) { if (tid < k) red[tid] += red[tid + k]; __syncthreads(); }
  if (tid == 0) out[0] = red[0] / (float)(B_ * (SC_ - 1));
}

// ---------------------------------------------------------------------------
// Host orchestration.
// Input layout (setup_inputs dict order; nested param dicts tree-flattened
// with alphabetical keys, each leaf stacked with leading L=6):
//  0 batch_eng  1 batch_chi  2 len_eng  3 len_chi  4 eng_emb  5 chi_emb
//  6..21  enc: an1_b,an1_g,an1_lb,an1_w, an2_b,an2_g,an2_lb,an2_w,
//              ff1_b,ff1_w,ff2_b,ff2_w, sa_kw,sa_ow,sa_qw,sa_vw
//  22..45 dec: an1_*, an2_*, an3_* (b,g,lb,w), ca_kw,ca_ow,ca_qw,ca_vw,
//              ff1_b,ff1_w,ff2_b,ff2_w, sa_kw,sa_ow,sa_qw,sa_vw
//  46 cls_w  47 cls_b
// Workspace: ~118 MB of scratch.
// ---------------------------------------------------------------------------
extern "C" void kernel_launch(void* const* d_in, const int* in_sizes, int n_in,
                              void* d_out, int out_size, void* d_ws, size_t ws_size,
                              hipStream_t stream)
{
  (void)in_sizes; (void)n_in; (void)out_size; (void)ws_size;

  const int*   batch_eng = (const int*)d_in[0];
  const int*   batch_chi = (const int*)d_in[1];
  const int*   len_eng   = (const int*)d_in[2];
  const int*   len_chi   = (const int*)d_in[3];
  const float* eng_emb   = (const float*)d_in[4];
  const float* chi_emb   = (const float*)d_in[5];
  const float* cls_w     = (const float*)d_in[46];
  const float* cls_b     = (const float*)d_in[47];

  const size_t DDsz = (size_t)D_ * D_;
  const size_t DFsz = (size_t)D_ * F_;

  // Workspace carve
  float* w = (float*)d_ws;
  size_t off = 0;
  const size_t RC = (size_t)B_ * SE_;   // 4096 rows (SE == SC)
  float* X  = w + off; off += RC * D_;  // encoder stream x (later: enc output)
  float* X1 = w + off; off += RC * D_;  // x1 / y1
  float* Y  = w + off; off += RC * D_;  // decoder stream y / y2
  float* Qb = w + off; off += RC * D_;
  float* Kb = w + off; off += RC * D_;
  float* Vb = w + off; off += RC * D_;
  float* Ob = w + off; off += RC * D_;
  float* T1 = w + off; off += RC * D_;
  float* T0 = w + off; off += RC * F_;  // FF intermediate / logits chunk
  float* ST = w + off; off += RC * 3;   // logsumexp state
  _Float16* Ah = (_Float16*)(w + off); off += RC * F_ / 2;  // f16 A mirror

  auto CVT = [&](const float* src, int n) {
    cvt_f32_f16<<<512, 256, 0, stream>>>(src, Ah, n);
  };
  auto GEMM = [&](const float* Win, const float* bin, float* Cout,
                  int N, int K, int ldw, int ldc, int relu) {
    dim3 g(N / 128, (int)RC / 128), blk(256);
    gemm_wmma<<<g, blk, 0, stream>>>(Ah, Win, bin, Cout, (int)RC, N, K, ldw, ldc, relu);
  };
  auto LN = [&](const float* tmp, const float* g_, const float* lb_,
                const float* res, float* out) {
    ln_add<<<(int)RC, 256, 0, stream>>>(tmp, g_, lb_, res, out);
  };
  auto PP = [&](int idx, int l, size_t per) -> const float* {
    return (const float*)d_in[idx] + (size_t)l * per;
  };
  const int ND = RC * D_, NF = RC * F_;

  // ------------------------- Embeddings -------------------------
  embed_pe<<<B_ * SE_, 256, 0, stream>>>(batch_eng, len_eng, eng_emb, X, SE_);
  embed_pe<<<B_ * SC_, 256, 0, stream>>>(batch_chi, len_chi, chi_emb, Y, SC_);

  // ------------------------- Encoder ----------------------------
  for (int l = 0; l < L_; ++l) {
    CVT(X, ND);
    GEMM(PP(20, l, DDsz), nullptr, Qb, D_, D_, D_, D_, 0);
    GEMM(PP(18, l, DDsz), nullptr, Kb, D_, D_, D_, D_, 0);
    GEMM(PP(21, l, DDsz), nullptr, Vb, D_, D_, D_, D_, 0);
    attn_fused<<<B_ * H_ * SE_ / 8, 256, 0, stream>>>(Qb, Kb, Vb, Ob, len_eng, SE_, SE_, 0);
    CVT(Ob, ND);
    GEMM(PP(19, l, DDsz), nullptr, T1, D_, D_, D_, D_, 0);              // a1
    CVT(T1, ND);
    GEMM(PP(9, l, DDsz), PP(6, l, D_), Qb, D_, D_, D_, D_, 0);          // a1@an1_w+b
    LN(Qb, PP(7, l, D_), PP(8, l, D_), X, X1);                          // x1 = LN + x
    CVT(X1, ND);
    GEMM(PP(15, l, DFsz), PP(14, l, F_), T0, F_, D_, F_, F_, 1);        // relu(ff1)
    CVT(T0, NF);
    GEMM(PP(17, l, DFsz), PP(16, l, D_), T1, D_, F_, D_, D_, 0);        // ff2
    CVT(T1, ND);
    GEMM(PP(13, l, DDsz), PP(10, l, D_), Qb, D_, D_, D_, D_, 0);        // f@an2_w+b
    LN(Qb, PP(11, l, D_), PP(12, l, D_), X1, X);                        // x = LN + x1
  }

  // ------------------------- Decoder ----------------------------
  for (int l = 0; l < L_; ++l) {
    // masked self-attention
    CVT(Y, ND);
    GEMM(PP(44, l, DDsz), nullptr, Qb, D_, D_, D_, D_, 0);
    GEMM(PP(42, l, DDsz), nullptr, Kb, D_, D_, D_, D_, 0);
    GEMM(PP(45, l, DDsz), nullptr, Vb, D_, D_, D_, D_, 0);
    attn_fused<<<B_ * H_ * SC_ / 8, 256, 0, stream>>>(Qb, Kb, Vb, Ob, len_chi, SC_, SC_, 1);
    CVT(Ob, ND);
    GEMM(PP(43, l, DDsz), nullptr, T1, D_, D_, D_, D_, 0);              // a1
    CVT(T1, ND);
    GEMM(PP(25, l, DDsz), PP(22, l, D_), Qb, D_, D_, D_, D_, 0);
    LN(Qb, PP(23, l, D_), PP(24, l, D_), T1, X1);                       // y1 = LN + a1
    // cross-attention (K/V from encoder output X)
    CVT(X1, ND);
    GEMM(PP(36, l, DDsz), nullptr, Qb, D_, D_, D_, D_, 0);
    CVT(X, ND);
    GEMM(PP(34, l, DDsz), nullptr, Kb, D_, D_, D_, D_, 0);
    GEMM(PP(37, l, DDsz), nullptr, Vb, D_, D_, D_, D_, 0);
    attn_fused<<<B_ * H_ * SC_ / 8, 256, 0, stream>>>(Qb, Kb, Vb, Ob, len_eng, SC_, SE_, 0);
    CVT(Ob, ND);
    GEMM(PP(35, l, DDsz), nullptr, T1, D_, D_, D_, D_, 0);              // c
    CVT(T1, ND);
    GEMM(PP(29, l, DDsz), PP(26, l, D_), Qb, D_, D_, D_, D_, 0);
    LN(Qb, PP(27, l, D_), PP(28, l, D_), X1, Y);                        // y2 = LN + y1
    // FFN
    CVT(Y, ND);
    GEMM(PP(39, l, DFsz), PP(38, l, F_), T0, F_, D_, F_, F_, 1);
    CVT(T0, NF);
    GEMM(PP(41, l, DFsz), PP(40, l, D_), T1, D_, F_, D_, D_, 0);
    CVT(T1, ND);
    GEMM(PP(33, l, DDsz), PP(30, l, D_), Qb, D_, D_, D_, D_, 0);
    LN(Qb, PP(31, l, D_), PP(32, l, D_), Y, Y);                         // y = LN + y2
  }

  // ----------------- Classifier + streaming NLL -----------------
  init_state<<<(B_ * SC_ + 255) / 256, 256, 0, stream>>>(ST);
  CVT(Y, ND);   // single f16 conversion reused by all 16 chunk GEMMs
  for (int c = 0; c * 2048 < VC_; ++c) {
    int Nc = VC_ - c * 2048;
    if (Nc > 2048) Nc = 2048;                 // chunks: 15x2048 + 1x1280
    GEMM(cls_w + (size_t)c * 2048, cls_b + (size_t)c * 2048, T0,
         Nc, D_, VC_, Nc, 0);
    lse_update<<<(int)RC, 256, 0, stream>>>(T0, ST, batch_chi, Nc, c * 2048, Nc);
  }
  nll_mean<<<1, 256, 0, stream>>>(ST, (float*)d_out);
}
